// WindowCrossAttention_22986664969101
// MI455X (gfx1250) — matrix operand order repaired and checked by reading, not compiled, not run on hardware
//
#include <hip/hip_runtime.h>
#include <math.h>

// ---------------------------------------------------------------------------
// MI455X (gfx1250) dual-stream cross-attention block, bf16 WMMA everywhere.
// wave32; WMMA shape 16x16x32 bf16 -> f32 acc.
// ---------------------------------------------------------------------------

typedef __bf16 bf16_t;
typedef __attribute__((ext_vector_type(16))) __bf16 v16bf;
typedef __attribute__((ext_vector_type(8)))  __bf16 v8bf;
typedef __attribute__((ext_vector_type(8)))  float  v8f;

union AFrag { v16bf v; v8bf h[2]; };

__device__ __forceinline__ v8f wmma_bf16(v16bf a, v16bf b, v8f c) {
  // 8 args: (neg_a, A, neg_b, B, c_mod, C, reuse_a, reuse_b)
  return __builtin_amdgcn_wmma_f32_16x16x32_bf16(false, a, false, b, (short)0, c,
                                                 false, false);
}

__device__ __forceinline__ float redmax16(float v) {
  v = fmaxf(v, __shfl_xor(v, 1));
  v = fmaxf(v, __shfl_xor(v, 2));
  v = fmaxf(v, __shfl_xor(v, 4));
  v = fmaxf(v, __shfl_xor(v, 8));
  return v;
}
__device__ __forceinline__ float redsum16(float v) {
  v += __shfl_xor(v, 1);
  v += __shfl_xor(v, 2);
  v += __shfl_xor(v, 4);
  v += __shfl_xor(v, 8);
  return v;
}
__device__ __forceinline__ float redsum32(float v) {
  v += __shfl_xor(v, 1);
  v += __shfl_xor(v, 2);
  v += __shfl_xor(v, 4);
  v += __shfl_xor(v, 8);
  v += __shfl_xor(v, 16);
  return v;
}

// ---------------------------------------------------------------------------
// Weight packing: y = x @ W^T, W stored [OUT][IN].  B[k][n] = W[n][k].
// Packed so a lane's 16 B-frag elements are contiguous:
//   packed[((kt*NT + nt)*32 + lane)*16 + e] = W[(nt*16+lo)*IN + kt*32 + hi*16 + e]
// ---------------------------------------------------------------------------
__global__ void pack_w(const float* __restrict__ W, bf16_t* __restrict__ P,
                       int IN, int OUT) {
  int idx = blockIdx.x * 256 + threadIdx.x;
  if (idx >= IN * OUT) return;
  int e    = idx & 15;
  int lane = (idx >> 4) & 31;
  int tile = idx >> 9;
  int NT = OUT >> 4;
  int kt = tile / NT, nt = tile - kt * NT;
  int lo = lane & 15, hi = lane >> 4;
  int k = kt * 32 + hi * 16 + e;
  int n = nt * 16 + lo;
  P[idx] = (bf16_t)W[(size_t)n * IN + k];
}

// ---------------------------------------------------------------------------
// LN1: per token (wave per token): write normalized bf16 + raw bf16 copy of x.
// ---------------------------------------------------------------------------
__global__ void ln1_kernel(const float* __restrict__ x0, const float* __restrict__ x1,
                           const float* __restrict__ w, const float* __restrict__ b,
                           bf16_t* __restrict__ nrm, bf16_t* __restrict__ xb) {
  const int lane = threadIdx.x & 31;
  const int t = blockIdx.x * 8 + (threadIdx.x >> 5);
  const float* x = (t >= 4096) ? (x1 + (size_t)(t - 4096) * 256)
                               : (x0 + (size_t)t * 256);
  float v[8];
  float s = 0.f;
#pragma unroll
  for (int i = 0; i < 8; i++) { v[i] = x[lane * 8 + i]; s += v[i]; }
  s = redsum32(s);
  float mean = s * (1.0f / 256.0f);
  float sq = 0.f;
#pragma unroll
  for (int i = 0; i < 8; i++) { float d = v[i] - mean; sq += d * d; }
  sq = redsum32(sq);
  float rstd = rsqrtf(sq * (1.0f / 256.0f) + 1e-5f);
#pragma unroll
  for (int i = 0; i < 8; i++) {
    int c = lane * 8 + i;
    float nv = (v[i] - mean) * rstd * w[c] + b[c];
    nrm[(size_t)t * 256 + c] = (bf16_t)nv;
    xb [(size_t)t * 256 + c] = (bf16_t)v[i];
  }
}

// ---------------------------------------------------------------------------
// Generic bf16 WMMA GEMM: out[M][N] = act(A[M][K] @ Bpacked * scale + bias).
// A split into two K-halves (A1: K1 cols, A2: K-K1 cols) to fuse the FFN
// concat([x, m]) without materializing it.  Wave computes 16x64 output.
// ---------------------------------------------------------------------------
__global__ void gemm_wmma(const bf16_t* __restrict__ A1, const bf16_t* __restrict__ A2,
                          int K1, const bf16_t* __restrict__ Bp,
                          const float* __restrict__ bias,
                          bf16_t* __restrict__ outb, float* __restrict__ outf,
                          int M, int N, int K, float scale, int gelu) {
  const int lane = threadIdx.x & 31;
  const int lo = lane & 15, hi = lane >> 4;
  const int task = blockIdx.x * (blockDim.x >> 5) + (threadIdx.x >> 5);
  const int nq_count = N >> 6;
  if (task >= (M >> 4) * nq_count) return;
  const int mt = task / nq_count;
  const int nq = task - mt * nq_count;
  const int NT = N >> 4;
  const int KT = K >> 5;
  const int row = mt * 16 + lo;
  const int K2 = K - K1;

  v8f acc0 = {0,0,0,0,0,0,0,0};
  v8f acc1 = acc0, acc2 = acc0, acc3 = acc0;

  for (int kt = 0; kt < KT; ++kt) {
    const int kcol = kt << 5;
    const bf16_t* pA = (kcol < K1)
        ? (A1 + (size_t)row * K1 + kcol)
        : (A2 + (size_t)row * K2 + (kcol - K1));
    AFrag a;
    a.h[0] = *(const v8bf*)(pA + hi * 16);
    a.h[1] = *(const v8bf*)(pA + hi * 16 + 8);

    const bf16_t* pB = Bp + (((size_t)kt * NT + (nq << 2)) * 32 + lane) * 16;
    v16bf b0 = *(const v16bf*)(pB);
    v16bf b1 = *(const v16bf*)(pB + 512);
    v16bf b2 = *(const v16bf*)(pB + 1024);
    v16bf b3 = *(const v16bf*)(pB + 1536);

    acc0 = wmma_bf16(a.v, b0, acc0);
    acc1 = wmma_bf16(a.v, b1, acc1);
    acc2 = wmma_bf16(a.v, b2, acc2);
    acc3 = wmma_bf16(a.v, b3, acc3);
  }

  v8f accs[4] = {acc0, acc1, acc2, acc3};
#pragma unroll
  for (int j = 0; j < 4; j++) {
    int n = ((nq << 2) + j) * 16 + lo;
    float bv = bias ? bias[n] : 0.0f;
#pragma unroll
    for (int r = 0; r < 8; r++) {
      int orow = mt * 16 + hi * 8 + r;
      float val = accs[j][r] * scale + bv;
      if (gelu) val = 0.5f * val * (1.0f + erff(val * 0.70710678118654752f));
      if (outb) outb[(size_t)orow * N + n] = (bf16_t)val;
      else      outf[(size_t)orow * N + n] = val;
    }
  }
}

// ---------------------------------------------------------------------------
// v transpose: vv[stream][b*2048+s][c] -> vT[(stream*2+b)*256 + c][s]
// so PV B-frags are contiguous 32B loads.
// ---------------------------------------------------------------------------
__global__ void transpose_v(const bf16_t* __restrict__ vv, bf16_t* __restrict__ vT) {
  int idx = blockIdx.x * 256 + threadIdx.x;   // 8192*256 elements
  int c = idx & 255;
  int t = idx >> 8;
  int s = t & 2047;
  int bidx = (t >> 11) & 3;                   // stream*2 + b
  vT[((size_t)bidx * 256 + c) * 2048 + s] = vv[idx];
}

// ---------------------------------------------------------------------------
// Flash cross-attention, both directions:
//   dir=0: m0 = softmax_row(qk0 qk1^T) v1     dir=1: m1 = softmax_row(qk1 qk0^T) v0
// One wave handles 16 query rows x all 2048 kv, online softmax, 32-kv steps.
// ---------------------------------------------------------------------------
__global__ void flash_attn(const bf16_t* __restrict__ qk, const bf16_t* __restrict__ vT,
                           bf16_t* __restrict__ attn) {
  __shared__ __attribute__((aligned(16))) bf16_t pbuf[8][16][40];
  const int lane = threadIdx.x & 31;
  const int lo = lane & 15, hi = lane >> 4;
  const int w = threadIdx.x >> 5;
  const int task = blockIdx.x * 8 + w;
  if (task >= 4096) return;
  const int qb  = task & 127;
  const int h   = (task >> 7) & 7;
  const int b   = (task >> 10) & 1;
  const int dir = (task >> 11) & 1;

  const size_t sstride = (size_t)4096 * 256;
  const bf16_t* Q  = qk + (size_t)dir * sstride + (size_t)b * 2048 * 256 + h * 32;
  const bf16_t* Kp = qk + (size_t)(1 - dir) * sstride + (size_t)b * 2048 * 256 + h * 32;
  const bf16_t* Vt = vT + (((size_t)(1 - dir) * 2 + b) * 256 + h * 32) * 2048;
  bf16_t* O = attn + (size_t)dir * sstride + (size_t)b * 2048 * 256 + h * 32;

  AFrag qf;
  {
    const bf16_t* pQ = Q + (size_t)(qb * 16 + lo) * 256;
    qf.h[0] = *(const v8bf*)(pQ + hi * 16);
    qf.h[1] = *(const v8bf*)(pQ + hi * 16 + 8);
  }

  float mrow[8], lsum[8];
#pragma unroll
  for (int r = 0; r < 8; r++) { mrow[r] = -1e30f; lsum[r] = 0.0f; }
  v8f O0 = {0,0,0,0,0,0,0,0};
  v8f O1 = O0;
  const v8f zc = O0;

  for (int s0 = 0; s0 < 2048; s0 += 32) {
    // S = Q K^T : B-frag rows come straight from row-major K (contiguous 32B)
    v16bf kb0 = *(const v16bf*)(Kp + (size_t)(s0 + lo) * 256 + hi * 16);
    v16bf kb1 = *(const v16bf*)(Kp + (size_t)(s0 + 16 + lo) * 256 + hi * 16);
    v8f sa = wmma_bf16(qf.v, kb0, zc);
    v8f sb = wmma_bf16(qf.v, kb1, zc);

#pragma unroll
    for (int r = 0; r < 8; r++) {
      float tm = redmax16(fmaxf(sa[r], sb[r]));
      float mn = fmaxf(mrow[r], tm);
      float corr = __expf(mrow[r] - mn);
      float p0 = __expf(sa[r] - mn);
      float p1 = __expf(sb[r] - mn);
      float ts = redsum16(p0 + p1);
      lsum[r] = lsum[r] * corr + ts;
      mrow[r] = mn;
      O0[r] *= corr;
      O1[r] *= corr;
      pbuf[w][hi * 8 + r][lo]      = (bf16_t)p0;
      pbuf[w][hi * 8 + r][16 + lo] = (bf16_t)p1;
    }
    // P (16x32) as A-frag via LDS bounce (same-wave DS ops are in order)
    AFrag pf;
    pf.h[0] = *(const v8bf*)&pbuf[w][lo][hi * 16];
    pf.h[1] = *(const v8bf*)&pbuf[w][lo][hi * 16 + 8];

    v16bf vb0 = *(const v16bf*)(Vt + (size_t)lo * 2048 + s0 + hi * 16);
    v16bf vb1 = *(const v16bf*)(Vt + (size_t)(16 + lo) * 2048 + s0 + hi * 16);
    O0 = wmma_bf16(pf.v, vb0, O0);
    O1 = wmma_bf16(pf.v, vb1, O1);
  }

#pragma unroll
  for (int r = 0; r < 8; r++) {
    float inv = 1.0f / lsum[r];
    size_t rowoff = (size_t)(qb * 16 + hi * 8 + r) * 256;
    O[rowoff + lo]      = (bf16_t)(O0[r] * inv);
    O[rowoff + 16 + lo] = (bf16_t)(O1[r] * inv);
  }
}

// ---------------------------------------------------------------------------
// LN2 + gamma + residual -> fp32 output.
// ---------------------------------------------------------------------------
__global__ void ln2_residual(const float* __restrict__ x0, const float* __restrict__ x1,
                             const float* __restrict__ h2,
                             const float* __restrict__ w, const float* __restrict__ b,
                             const float* __restrict__ gamma,
                             float* __restrict__ out) {
  const int lane = threadIdx.x & 31;
  const int t = blockIdx.x * 8 + (threadIdx.x >> 5);
  const float* x = (t >= 4096) ? (x1 + (size_t)(t - 4096) * 256)
                               : (x0 + (size_t)t * 256);
  const float* hrow = h2 + (size_t)t * 256;
  float v[8];
  float s = 0.f;
#pragma unroll
  for (int i = 0; i < 8; i++) { v[i] = hrow[lane * 8 + i]; s += v[i]; }
  s = redsum32(s);
  float mean = s * (1.0f / 256.0f);
  float sq = 0.f;
#pragma unroll
  for (int i = 0; i < 8; i++) { float d = v[i] - mean; sq += d * d; }
  sq = redsum32(sq);
  float rstd = rsqrtf(sq * (1.0f / 256.0f) + 1e-5f);
#pragma unroll
  for (int i = 0; i < 8; i++) {
    int c = lane * 8 + i;
    float ln = (v[i] - mean) * rstd * w[c] + b[c];
    out[(size_t)t * 256 + c] = x[c] + gamma[c] * ln;
  }
}

// ---------------------------------------------------------------------------
extern "C" void kernel_launch(void* const* d_in, const int* in_sizes, int n_in,
                              void* d_out, int out_size, void* d_ws, size_t ws_size,
                              hipStream_t stream) {
  const float* x0      = (const float*)d_in[0];
  const float* x1      = (const float*)d_in[1];
  const float* qk_w    = (const float*)d_in[2];
  const float* v_w     = (const float*)d_in[3];
  const float* merge_w = (const float*)d_in[4];
  const float* ln1_w   = (const float*)d_in[5];
  const float* ln1_b   = (const float*)d_in[6];
  const float* ln2_w   = (const float*)d_in[7];
  const float* ln2_b   = (const float*)d_in[8];
  const float* fc1_w   = (const float*)d_in[9];
  const float* fc1_b   = (const float*)d_in[10];
  const float* fc2_w   = (const float*)d_in[11];
  const float* fc2_b   = (const float*)d_in[12];
  const float* gamma   = (const float*)d_in[13];
  float* outp = (float*)d_out;

  char* ws = (char*)d_ws;
  const size_t MB = (size_t)1 << 20;
  bf16_t* nrm  = (bf16_t*)(ws + 0 * MB);      // [8192][256] bf16
  bf16_t* xb   = (bf16_t*)(ws + 4 * MB);      // raw x as bf16
  bf16_t* qkb  = (bf16_t*)(ws + 8 * MB);      // scaled qk projection
  bf16_t* vv   = (bf16_t*)(ws + 12 * MB);     // v projection
  bf16_t* vT   = (bf16_t*)(ws + 16 * MB);     // v transposed per (stream,b)
  bf16_t* att  = (bf16_t*)(ws + 20 * MB);     // attention output (unheaded)
  bf16_t* mrg  = (bf16_t*)(ws + 24 * MB);     // after merge_w
  bf16_t* h1   = (bf16_t*)(ws + 28 * MB);     // [8192][1024] after GELU
  float*  h2   = (float*) (ws + 44 * MB);     // [8192][256] fp32 after fc2
  bf16_t* wpqk = (bf16_t*)(ws + 52 * MB);
  bf16_t* wpv  = (bf16_t*)(ws + 52 * MB + 128 * 1024);
  bf16_t* wpmg = (bf16_t*)(ws + 52 * MB + 256 * 1024);
  bf16_t* wpf1 = (bf16_t*)(ws + 52 * MB + 384 * 1024);
  bf16_t* wpf2 = (bf16_t*)(ws + 53 * MB + 384 * 1024);

  // 1. pack weights (bf16, B-frag tile layout)
  pack_w<<<(256 * 256 + 255) / 256, 256, 0, stream>>>(qk_w,    wpqk, 256, 256);
  pack_w<<<(256 * 256 + 255) / 256, 256, 0, stream>>>(v_w,     wpv,  256, 256);
  pack_w<<<(256 * 256 + 255) / 256, 256, 0, stream>>>(merge_w, wpmg, 256, 256);
  pack_w<<<(512 * 1024 + 255) / 256, 256, 0, stream>>>(fc1_w,  wpf1, 512, 1024);
  pack_w<<<(1024 * 256 + 255) / 256, 256, 0, stream>>>(fc2_w,  wpf2, 1024, 256);

  // 2. LN1 + bf16 casts
  ln1_kernel<<<1024, 256, 0, stream>>>(x0, x1, ln1_w, ln1_b, nrm, xb);

  // 3. qk / v projections (M=8192, N=256, K=256); scale = 32^-0.25 baked into qk
  const float qscale = 0.42044820762685725f;
  gemm_wmma<<<256, 256, 0, stream>>>(nrm, nrm, 256, wpqk, nullptr, qkb, nullptr,
                                     8192, 256, 256, qscale, 0);
  gemm_wmma<<<256, 256, 0, stream>>>(nrm, nrm, 256, wpv, nullptr, vv, nullptr,
                                     8192, 256, 256, 1.0f, 0);

  // 4. v transpose for contiguous PV B-frags
  transpose_v<<<8192, 256, 0, stream>>>(vv, vT);

  // 5. flash cross-attention, both softmax directions
  flash_attn<<<512, 256, 0, stream>>>(qkb, vT, att);

  // 6. merge projection
  gemm_wmma<<<256, 256, 0, stream>>>(att, att, 256, wpmg, nullptr, mrg, nullptr,
                                     8192, 256, 256, 1.0f, 0);

  // 7. fc1 on concat([x, m]) via split-K (K1=256 from xb, rest from mrg) + GELU
  gemm_wmma<<<1024, 256, 0, stream>>>(xb, mrg, 256, wpf1, fc1_b, h1, nullptr,
                                      8192, 1024, 512, 1.0f, 1);

  // 8. fc2 -> fp32
  gemm_wmma<<<256, 256, 0, stream>>>(h1, h1, 1024, wpf2, fc2_b, nullptr, h2,
                                     8192, 256, 1024, 1.0f, 0);

  // 9. LN2 + gamma + residual -> d_out
  ln2_residual<<<1024, 256, 0, stream>>>(x0, x1, h2, ln2_w, ln2_b, gamma, outp);
}
